// GeneralSeq2Seq_85633057947931
// MI455X (gfx1250) — matrix operand order repaired and checked by reading, not compiled, not run on hardware
//
#include <hip/hip_runtime.h>
#include <math.h>

#define B_   128
#define T_   2048
#define F1_  15
#define FC_  8
#define H_   256
#define G4_  1024           // 4*H
#define KE_  288            // padded encoder K: 16 + 256 -> 9*32
#define KD_  544            // padded decoder K: 257 + 256 -> 17*32
#define TRG_ 24

typedef __bf16 bf16;
typedef __attribute__((ext_vector_type(16))) __bf16 v16bf;
typedef __attribute__((ext_vector_type(8)))  __bf16 v8bf;
typedef __attribute__((ext_vector_type(8)))  float  v8f;

__device__ __forceinline__ unsigned short f2bf(float f) {
  union { float f; unsigned u; } v; v.f = f;
  unsigned r = v.u + 0x7fffu + ((v.u >> 16) & 1u);
  return (unsigned short)(r >> 16);
}
__device__ __forceinline__ float bf2f(unsigned short s) {
  union { unsigned u; float f; } v; v.u = ((unsigned)s) << 16; return v.f;
}
__device__ __forceinline__ float sigm(float x) { return 1.0f / (1.0f + __expf(-x)); }
__device__ __forceinline__ float tanh_f(float x) {
  float e = __expf(2.0f * x);
  return (e - 1.0f) / (e + 1.0f);
}
// Defeat loop-invariant hoisting of the weight-fragment loads WITHOUT destroying
// address-space provenance: launder a zero offset (not the pointer) through an
// opaque volatile asm. Loads stay GLOBAL (saddr form), but cannot be hoisted out
// of the recurrence loop into 288 spilled VGPRs.
__device__ __forceinline__ const unsigned short* keep_in_loop(const unsigned short* p) {
  unsigned off = 0;
  __asm__ volatile("" : "+s"(off));
  return p + off;
}

// ---- WMMA fragment loaders (contiguous k order shared by A and B, wave32) ----
// A (16xK tile, row m = lane&15, half h = lane>>4):
//   elem e -> K = k0 + 16h + e
__device__ __forceinline__ v16bf loadA16(const unsigned short* base, int ld, int lane, int k0) {
  int m = lane & 15, h = lane >> 4;
  const unsigned short* r = base + (size_t)m * ld + k0 + 16 * h;
  union { v16bf v; v8bf p[2]; } u;
  u.p[0] = *(const v8bf*)(r);
  u.p[1] = *(const v8bf*)(r + 8);
  return u.v;
}
// B (Kx16 tile from row-major W[n][k]): lane col n = lane&15, half h = lane>>4,
//   elem e -> K = k0 + 16h + e   (one contiguous 32-byte load)
__device__ __forceinline__ v16bf loadB16(const unsigned short* W, int ld, int ncol, int lane, int k0) {
  int n = lane & 15, h = lane >> 4;
  return *(const v16bf*)(W + (size_t)(ncol + n) * ld + k0 + 16 * h);
}
__device__ __forceinline__ v8f wmma_bf16(v16bf a, v16bf b, v8f c) {
  return __builtin_amdgcn_wmma_f32_16x16x32_bf16(false, a, false, b, (short)0, c, false, false);
}

// ================= prep: pack x = [src1 | src2@Wc+bc] as bf16, zero eproj =================
__global__ void prep_xcat_kernel(const float* __restrict__ src1, const float* __restrict__ src2,
                                 const float* __restrict__ Wc, const float* __restrict__ bc,
                                 unsigned short* __restrict__ xcat, float* __restrict__ eproj) {
  size_t i = (size_t)blockIdx.x * blockDim.x + threadIdx.x;
  if (i >= (size_t)B_ * T_) return;
  float sm = bc[0];
#pragma unroll
  for (int f = 0; f < FC_; ++f) sm += src2[i * FC_ + f] * Wc[f];
#pragma unroll
  for (int f = 0; f < F1_; ++f) xcat[i * 16 + f] = f2bf(src1[i * F1_ + f]);
  xcat[i * 16 + 15] = f2bf(sm);
  eproj[i] = 0.0f;
}

// ================= prep: bf16 weight packing (concat + K-pad, row-major [n][k]) ============
__global__ void prep_w_kernel(const float* __restrict__ Wih_e, const float* __restrict__ Whh_e,
                              const float* __restrict__ bih_e, const float* __restrict__ bhh_e,
                              const float* __restrict__ Wfc,   const float* __restrict__ Wih_d,
                              const float* __restrict__ Whh_d, const float* __restrict__ bih_d,
                              const float* __restrict__ bhh_d,
                              unsigned short* __restrict__ WcatE, float* __restrict__ be,
                              unsigned short* __restrict__ WfcB,
                              unsigned short* __restrict__ WcatD, float* __restrict__ bd) {
  size_t stride = (size_t)gridDim.x * blockDim.x;
  size_t t0 = (size_t)blockIdx.x * blockDim.x + threadIdx.x;
  for (size_t i = t0; i < (size_t)G4_ * KE_; i += stride) {
    int nn = (int)(i / KE_), k = (int)(i % KE_);
    float v = (k < 16) ? Wih_e[nn * 16 + k] : (k < 272 ? Whh_e[(size_t)nn * H_ + (k - 16)] : 0.0f);
    WcatE[i] = f2bf(v);
  }
  for (size_t i = t0; i < (size_t)G4_ * KD_; i += stride) {
    int nn = (int)(i / KD_), k = (int)(i % KD_);
    float v = (k < 257) ? Wih_d[(size_t)nn * 257 + k]
                        : (k < 513 ? Whh_d[(size_t)nn * H_ + (k - 257)] : 0.0f);
    WcatD[i] = f2bf(v);
  }
  for (size_t i = t0; i < (size_t)H_ * H_; i += stride) WfcB[i] = f2bf(Wfc[i]);
  for (size_t i = t0; i < (size_t)G4_; i += stride) {
    be[i] = bih_e[i] + bhh_e[i];
    bd[i] = bih_d[i] + bhh_d[i];
  }
}

// ================= encoder LSTM recurrence: 8 WGs x 16 batch rows, WMMA per step ==========
__global__ __launch_bounds__(512) void enc_lstm_kernel(
    const unsigned short* __restrict__ xcat,   // (B,T,16) bf16
    const unsigned short* __restrict__ Wcat,   // (1024,288) bf16 [n][k], L2-resident
    const float* __restrict__ be,              // (1024)
    unsigned short* __restrict__ outs,         // (B,T,256) bf16
    float* __restrict__ c_fin)                 // (B,256) final cell state
{
  __shared__ unsigned short Xs[16 * KE_];      // [x_t(16) | h(256) | pad(16)] per row
  const int tid = threadIdx.x;
  const int lane = tid & 31, wid = tid >> 5;   // 16 waves
  const int b0 = blockIdx.x * 16;
  const int n = lane & 15, h = lane >> 4;
  const int hid = wid * 16 + n;                // hidden slice owned by this wave/lane

  float bq[4];
#pragma unroll
  for (int q = 0; q < 4; ++q) bq[q] = be[q * H_ + hid];

  for (int i = tid; i < 16 * KE_; i += 512) Xs[i] = 0;
  __syncthreads();
  if (tid < 256) {
    int m = tid >> 4, col = tid & 15;
    Xs[m * KE_ + col] = xcat[((size_t)(b0 + m) * T_ + 0) * 16 + col];
  }
  __syncthreads();

  v8f c = {0.f, 0.f, 0.f, 0.f, 0.f, 0.f, 0.f, 0.f};

#pragma unroll 1
  for (int t = 0; t < T_; ++t) {
    // weights re-read from L2 every step (do NOT let LICM hoist 288 regs of B)
    const unsigned short* Wt = keep_in_loop(Wcat);

    v16bf a[9];
#pragma unroll
    for (int kc = 0; kc < 9; ++kc) a[kc] = loadA16(Xs, KE_, lane, kc * 32);

    v8f acc[4];
#pragma unroll
    for (int q = 0; q < 4; ++q) {
      v8f d = {0.f, 0.f, 0.f, 0.f, 0.f, 0.f, 0.f, 0.f};
      int ncol = q * H_ + wid * 16;
#pragma unroll
      for (int kc = 0; kc < 9; ++kc)
        d = wmma_bf16(a[kc], loadB16(Wt, KE_, ncol, lane, kc * 32), d);
      acc[q] = d;
    }

    unsigned short hv[8];
#pragma unroll
    for (int j = 0; j < 8; ++j) {
      float iv = sigm(acc[0][j] + bq[0]);
      float fv = sigm(acc[1][j] + bq[1]);
      float gv = tanh_f(acc[2][j] + bq[2]);
      float ov = sigm(acc[3][j] + bq[3]);
      float cj = fv * c[j] + iv * gv;
      c[j] = cj;
      hv[j] = f2bf(ov * tanh_f(cj));
    }

    // ---- split barrier #1: everyone done READING Xs; overlap global h stores ----
    __asm__ volatile("s_wait_dscnt 0x0" ::: "memory");
    __asm__ volatile("s_barrier_signal -1" ::: "memory");
#pragma unroll
    for (int j = 0; j < 8; ++j)
      outs[((size_t)(b0 + j + 8 * h) * T_ + t) * H_ + hid] = hv[j];
    __asm__ volatile("s_barrier_wait -1" ::: "memory");

    // ---- write phase: new h and x_{t+1} into Xs ----
#pragma unroll
    for (int j = 0; j < 8; ++j) Xs[(j + 8 * h) * KE_ + 16 + hid] = hv[j];
    if (t + 1 < T_ && tid < 256) {
      int m = tid >> 4, col = tid & 15;
      const unsigned short* gx = xcat + ((size_t)(b0 + m) * T_ + (t + 1)) * 16 + col;
      Xs[m * KE_ + col] = *gx;
      if (t + 2 < T_) __builtin_prefetch(gx + 16, 0, 3);  // global_prefetch_b8, next row
    }

    // ---- split barrier #2: new [x|h] visible ----
    __asm__ volatile("s_wait_dscnt 0x0" ::: "memory");
    __asm__ volatile("s_barrier_signal -1" ::: "memory");
    __asm__ volatile("s_barrier_wait -1" ::: "memory");
  }
#pragma unroll
  for (int j = 0; j < 8; ++j)
    c_fin[(size_t)(b0 + j + 8 * h) * H_ + hid] = c[j];
}

// ================= enc = outs @ Wfc.T + bfc (WMMA GEMM), fused enc@Wa_e -> eproj ==========
__global__ __launch_bounds__(256) void enc_fc_kernel(
    const unsigned short* __restrict__ outs,  // (B*T,256) bf16
    const unsigned short* __restrict__ WfcB,  // (256,256) bf16 [n][k]
    const float* __restrict__ bfc,            // (256)
    const float* __restrict__ Wa,             // (512): [256..511] = Wa_e
    unsigned short* __restrict__ enc,         // (B*T,256) bf16
    float* __restrict__ eproj)                // (B*T)
{
  const int tid = threadIdx.x, lane = tid & 31, wid = tid >> 5;
  const size_t r0 = (size_t)blockIdx.x * 16;
  const int n = lane & 15, h = lane >> 4;

  v16bf a[8];
#pragma unroll
  for (int kc = 0; kc < 8; ++kc) a[kc] = loadA16(outs + r0 * H_, H_, lane, kc * 32);

#pragma unroll
  for (int cti = 0; cti < 2; ++cti) {
    int ncol = (wid + cti * 8) * 16;
    v8f d = {0.f, 0.f, 0.f, 0.f, 0.f, 0.f, 0.f, 0.f};
#pragma unroll
    for (int kc = 0; kc < 8; ++kc)
      d = wmma_bf16(a[kc], loadB16(WfcB, H_, ncol, lane, kc * 32), d);

    float bv = bfc[ncol + n];
    float wae = Wa[256 + ncol + n];
#pragma unroll
    for (int j = 0; j < 8; ++j) {
      float v = d[j] + bv;
      enc[(r0 + j + 8 * h) * H_ + ncol + n] = f2bf(v);
      float p = v * wae;
      p += __shfl_xor(p, 1, 32);
      p += __shfl_xor(p, 2, 32);
      p += __shfl_xor(p, 4, 32);
      p += __shfl_xor(p, 8, 32);
      if (n == 0) atomicAdd(&eproj[r0 + j + 8 * h], p);
    }
  }
}

// ================= decoder init: token = enc[:,-1,:]@Wf2+bf2; X2 h <- final enc h ==========
__global__ __launch_bounds__(256) void dec_init_kernel(
    const unsigned short* __restrict__ outs, const unsigned short* __restrict__ enc,
    const float* __restrict__ Wf2, const float* __restrict__ bf2,
    unsigned short* __restrict__ X2)  // (B,544): [inp | ctx(256) | h(256) | pad(31)]
{
  int b = blockIdx.x, tid = threadIdx.x;
  __shared__ float red[256];
  size_t last = ((size_t)b * T_ + (T_ - 1)) * H_;
  X2[(size_t)b * KD_ + 257 + tid] = outs[last + tid];
  if (tid < KD_ - 513) X2[(size_t)b * KD_ + 513 + tid] = 0;
  red[tid] = bf2f(enc[last + tid]) * Wf2[tid];
  __syncthreads();
  for (int s = 128; s > 0; s >>= 1) { if (tid < s) red[tid] += red[tid + s]; __syncthreads(); }
  if (tid == 0) X2[(size_t)b * KD_] = f2bf(red[0] + bf2[0]);
}

// ================= decoder attention: softmax(tanh(h@Wa_h + eproj)) weighted sum of enc ===
__global__ __launch_bounds__(256) void dec_attn_kernel(
    const unsigned short* __restrict__ enc, const float* __restrict__ eproj,
    const float* __restrict__ Wa,  // [0..255] = Wa_h
    unsigned short* __restrict__ X2)
{
  int b = blockIdx.x, tid = threadIdx.x;
  __shared__ float earr[T_];
  __shared__ float red[256];

  red[tid] = bf2f(X2[(size_t)b * KD_ + 257 + tid]) * Wa[tid];
  __syncthreads();
  for (int s = 128; s > 0; s >>= 1) { if (tid < s) red[tid] += red[tid + s]; __syncthreads(); }
  float ha = red[0];
  __syncthreads();

  float lm = -1e30f;
  for (int t = tid; t < T_; t += 256) {
    float e = tanh_f(ha + eproj[(size_t)b * T_ + t]);
    earr[t] = e;
    lm = fmaxf(lm, e);
  }
  red[tid] = lm;
  __syncthreads();
  for (int s = 128; s > 0; s >>= 1) { if (tid < s) red[tid] = fmaxf(red[tid], red[tid + s]); __syncthreads(); }
  float mx = red[0];
  __syncthreads();

  float ls = 0.f;
  for (int t = tid; t < T_; t += 256) {
    float w = __expf(earr[t] - mx);
    earr[t] = w;
    ls += w;
  }
  red[tid] = ls;
  __syncthreads();
  for (int s = 128; s > 0; s >>= 1) { if (tid < s) red[tid] += red[tid + s]; __syncthreads(); }
  float inv = 1.0f / red[0];
  __syncthreads();

  float acc = 0.f;
  const unsigned short* ep = enc + (size_t)b * T_ * H_ + tid;
#pragma unroll 4
  for (int t = 0; t < T_; ++t) acc += earr[t] * bf2f(ep[(size_t)t * H_]);
  X2[(size_t)b * KD_ + 1 + tid] = f2bf(acc * inv);
}

// ================= decoder LSTM cell (single step, WMMA) ==================================
__global__ __launch_bounds__(512) void dec_lstm_kernel(
    const unsigned short* __restrict__ WcatD,  // (1024,544) bf16 [n][k]
    const float* __restrict__ bd,              // (1024)
    unsigned short* __restrict__ X2,           // (B,544)
    float* __restrict__ cst)                   // (B,256)
{
  __shared__ unsigned short Xs[16 * KD_];
  const int tid = threadIdx.x, lane = tid & 31, wid = tid >> 5;
  const int b0 = blockIdx.x * 16;
  const int n = lane & 15, h = lane >> 4;
  const int hid = wid * 16 + n;

  for (int i = tid; i < 16 * KD_; i += 512)
    Xs[i] = X2[(size_t)(b0 + i / KD_) * KD_ + (i % KD_)];
  __syncthreads();

  v8f acc[4];
#pragma unroll
  for (int q = 0; q < 4; ++q) {
    v8f d = {0.f, 0.f, 0.f, 0.f, 0.f, 0.f, 0.f, 0.f};
    int ncol = q * H_ + wid * 16;
#pragma unroll
    for (int kc = 0; kc < 17; ++kc)
      d = wmma_bf16(loadA16(Xs, KD_, lane, kc * 32),
                    loadB16(WcatD, KD_, ncol, lane, kc * 32), d);
    acc[q] = d;
  }
#pragma unroll
  for (int j = 0; j < 8; ++j) {
    int row = b0 + j + 8 * h;
    float iv = sigm(acc[0][j] + bd[0 * H_ + hid]);
    float fv = sigm(acc[1][j] + bd[1 * H_ + hid]);
    float gv = tanh_f(acc[2][j] + bd[2 * H_ + hid]);
    float ov = sigm(acc[3][j] + bd[3 * H_ + hid]);
    size_t ci = (size_t)row * H_ + hid;
    float cj = fv * cst[ci] + iv * gv;
    cst[ci] = cj;
    X2[(size_t)row * KD_ + 257 + hid] = f2bf(ov * tanh_f(cj));
  }
}

// ================= decoder output head: pred = h@Wo + bo; feed back as next inp ===========
__global__ __launch_bounds__(256) void dec_pred_kernel(
    const unsigned short* __restrict__ X2, const float* __restrict__ Wo,
    const float* __restrict__ bo, float* __restrict__ out,
    unsigned short* __restrict__ X2w, int step)
{
  int b = blockIdx.x, tid = threadIdx.x;
  __shared__ float red[256];
  red[tid] = bf2f(X2[(size_t)b * KD_ + 257 + tid]) * Wo[tid];
  __syncthreads();
  for (int s = 128; s > 0; s >>= 1) { if (tid < s) red[tid] += red[tid + s]; __syncthreads(); }
  if (tid == 0) {
    float pr = red[0] + bo[0];
    out[(size_t)b * TRG_ + step] = pr;
    X2w[(size_t)b * KD_] = f2bf(pr);
  }
}

extern "C" void kernel_launch(void* const* d_in, const int* in_sizes, int n_in,
                              void* d_out, int out_size, void* d_ws, size_t ws_size,
                              hipStream_t stream) {
  const float* src1  = (const float*)d_in[0];
  const float* src2  = (const float*)d_in[1];
  const float* Wc    = (const float*)d_in[2];
  const float* bc    = (const float*)d_in[3];
  const float* Wih_e = (const float*)d_in[4];
  const float* Whh_e = (const float*)d_in[5];
  const float* bih_e = (const float*)d_in[6];
  const float* bhh_e = (const float*)d_in[7];
  const float* Wfc_e = (const float*)d_in[8];
  const float* bfc_e = (const float*)d_in[9];
  const float* Wf2   = (const float*)d_in[10];
  const float* bf2   = (const float*)d_in[11];
  const float* Wa    = (const float*)d_in[12];
  const float* Wih_d = (const float*)d_in[13];
  const float* Whh_d = (const float*)d_in[14];
  const float* bih_d = (const float*)d_in[15];
  const float* bhh_d = (const float*)d_in[16];
  const float* Wo    = (const float*)d_in[17];
  const float* bo    = (const float*)d_in[18];
  (void)in_sizes; (void)n_in; (void)out_size; (void)ws_size;

  char* ws = (char*)d_ws;
  size_t off = 0;
  auto alloc = [&](size_t bytes) -> char* {
    char* p = ws + off;
    off = (off + bytes + 255) & ~(size_t)255;
    return p;
  };
  unsigned short* xcat  = (unsigned short*)alloc((size_t)B_ * T_ * 16 * 2);   // 8 MiB
  unsigned short* WcatE = (unsigned short*)alloc((size_t)G4_ * KE_ * 2);
  float*          be    = (float*)alloc((size_t)G4_ * 4);
  unsigned short* outs  = (unsigned short*)alloc((size_t)B_ * T_ * H_ * 2);   // 128 MiB
  unsigned short* enc   = (unsigned short*)alloc((size_t)B_ * T_ * H_ * 2);   // 128 MiB
  float*          eproj = (float*)alloc((size_t)B_ * T_ * 4);                 // 1 MiB
  unsigned short* WfcB  = (unsigned short*)alloc((size_t)H_ * H_ * 2);
  unsigned short* WcatD = (unsigned short*)alloc((size_t)G4_ * KD_ * 2);
  float*          bd    = (float*)alloc((size_t)G4_ * 4);
  unsigned short* X2    = (unsigned short*)alloc((size_t)B_ * KD_ * 2);
  float*          cst   = (float*)alloc((size_t)B_ * H_ * 4);

  prep_xcat_kernel<<<(B_ * T_ + 255) / 256, 256, 0, stream>>>(src1, src2, Wc, bc, xcat, eproj);
  prep_w_kernel<<<1024, 256, 0, stream>>>(Wih_e, Whh_e, bih_e, bhh_e, Wfc_e,
                                          Wih_d, Whh_d, bih_d, bhh_d,
                                          WcatE, be, WfcB, WcatD, bd);
  enc_lstm_kernel<<<B_ / 16, 512, 0, stream>>>(xcat, WcatE, be, outs, cst);
  enc_fc_kernel<<<B_ * T_ / 16, 256, 0, stream>>>(outs, WfcB, bfc_e, Wa, enc, eproj);
  dec_init_kernel<<<B_, 256, 0, stream>>>(outs, enc, Wf2, bf2, X2);
  for (int s = 0; s < TRG_; ++s) {
    dec_attn_kernel<<<B_, 256, 0, stream>>>(enc, eproj, Wa, X2);
    dec_lstm_kernel<<<B_ / 16, 512, 0, stream>>>(WcatD, bd, X2, cst);
    dec_pred_kernel<<<B_, 256, 0, stream>>>(X2, Wo, bo, (float*)d_out, X2, s);
  }
}
